// MIMO_85023172591646
// MI455X (gfx1250) — compile-verified, matrix operand order checked by reading, not run on hardware
//
#include <hip/hip_runtime.h>
#include <math.h>

#define DEVINL __device__ __forceinline__

typedef float v8f __attribute__((ext_vector_type(8)));
typedef __bf16 v16bf __attribute__((ext_vector_type(16)));
typedef unsigned int u32x4 __attribute__((ext_vector_type(4)));

constexpr int Bn  = 4096;   // batch
constexpr int Hn  = 128;    // k_state
constexpr int G3  = 384;    // 3*H
constexpr int KIN = 8;
constexpr int KOUT = 4;
constexpr int PRE = 64;
constexpr int Ln  = 128;
constexpr int Sn  = Bn * KIN;    // 32768, spline plane stride
constexpr float HS  = 0.1f;
constexpr float UTC = 0.1f;
constexpr int BT  = 64;          // batch rows per block

DEVINL unsigned short f2bf(float f) {
  union { float f; unsigned int u; } c; c.f = f;
  unsigned int u = c.u;
  unsigned int r = u + 0x7FFFu + ((u >> 16) & 1u);
  return (unsigned short)(r >> 16);
}

// Native TRANS-pipe tanh/exp/rcp where available (CDNA5 has V_TANH_F32 /
// V_EXP_F32 / V_RCP_F32 on the transcendental pipe, co-executing with VALU).
DEVINL float fast_tanh(float x) {
#if __has_builtin(__builtin_amdgcn_tanhf)
  return __builtin_amdgcn_tanhf(x);
#elif __has_builtin(__builtin_amdgcn_tanh_f32)
  return __builtin_amdgcn_tanh_f32(x);
#else
  return tanhf(x);
#endif
}

DEVINL float fast_sigmoid(float x) {
  return __builtin_amdgcn_rcpf(1.0f + __expf(-x));
}

DEVINL v8f vzero8() { v8f z = {0.f,0.f,0.f,0.f,0.f,0.f,0.f,0.f}; return z; }

union FragU { u32x4 q[2]; v16bf v; };

// Load one 16x32 bf16 WMMA operand fragment from a row-major [rows][ld] bf16
// array. Lanes 0-15: row M=lane, K = {kc*32+0..7, kc*32+16..23};
// lanes 16-31: same rows, K = {+8..15, +24..31}. Two 16B loads per lane.
DEVINL v16bf load_frag(const unsigned short* base, int ld, int row0, int kc, int lane) {
  int r = row0 + (lane & 15);
  int half = (lane >> 4) & 1;
  const unsigned short* p = base + r * ld + kc * 32 + half * 8;
  FragU u;
  u.q[0] = *(const u32x4*)(p);
  u.q[1] = *(const u32x4*)(p + 16);
  return u.v;
}

DEVINL v8f wmma_bf16(v16bf a, v16bf b, v8f c) {
  return __builtin_amdgcn_wmma_f32_16x16x32_bf16(false, a, false, b, (short)0, c, false, false);
}

DEVINL void copy_b16(unsigned short* dst, const unsigned short* src, int n16) {
  u32x4* d = (u32x4*)dst;
  const u32x4* s = (const u32x4*)src;
  int n4 = n16 >> 3;
  for (int i = threadIdx.x; i < n4; i += blockDim.x) d[i] = s[i];
}

DEVINL void copy_f32(float* dst, const float* src, int n) {
  for (int i = threadIdx.x; i < n; i += blockDim.x) dst[i] = src[i];
}

// ---------------- fp32 -> bf16 weight conversion ----------------
__global__ __launch_bounds__(256) void cvt_bf16(const float* __restrict__ src,
                                                unsigned short* __restrict__ dst, int n) {
  int i = blockIdx.x * blockDim.x + threadIdx.x;
  if (i < n) dst[i] = f2bf(src[i]);
}

// ---------------- natural cubic spline: Thomas solve ----------------
// Solves (4I + offdiag) Minner = rhs for 32768 independent columns.
// Mbuf layout: [Ln+1][Bn*KIN] fp32, Mbuf[0]=Mbuf[Ln]=0.
__global__ __launch_bounds__(256) void spline_kernel(const float* __restrict__ pre_x,
                                                     const float* __restrict__ forward_x,
                                                     float* __restrict__ Mbuf) {
  __shared__ float cps[Ln - 1]; // c' coefficients, data independent
  if (threadIdx.x == 0) {
    float c = 0.25f;
    cps[0] = c;
    for (int i = 1; i < Ln - 1; ++i) { c = 1.0f / (4.0f - c); cps[i] = c; }
  }
  __syncthreads();
  int q = blockIdx.x * blockDim.x + threadIdx.x; // 0..Sn-1
  const float kf = 6.0f / (HS * HS);
  // ys(i): i==0 -> pre_x[63] plane, else forward_x[i-1] plane
  float y0 = pre_x[63 * Sn + q];
  float y1 = forward_x[q];
  float dprev = 0.f;
  for (int i = 0; i < Ln - 1; ++i) {
    float y2 = forward_x[(i + 1) * Sn + q];
    float rhs = kf * (y0 - 2.0f * y1 + y2);
    dprev = (i == 0) ? rhs * 0.25f : (rhs - dprev) * cps[i];
    Mbuf[(1 + i) * Sn + q] = dprev;
    y0 = y1; y1 = y2;
  }
  float xn = Mbuf[(Ln - 1) * Sn + q]; // x_{nin-1} = d'_{nin-1}
  for (int i = Ln - 3; i >= 0; --i) {
    float xi = Mbuf[(1 + i) * Sn + q] - cps[i] * xn;
    Mbuf[(1 + i) * Sn + q] = xi;
    xn = xi;
  }
  Mbuf[q] = 0.0f;
  Mbuf[Ln * Sn + q] = 0.0f;
}

// ---------------- GRU encoder ----------------
// Block = 64 batch rows, 8 waves = 4 row-tiles x 2 col-groups (64 gate cols each).
__global__ __launch_bounds__(256) void encoder_kernel(
    const float* __restrict__ pre_x, const float* __restrict__ pre_y,
    const float* __restrict__ W_ih, const float* __restrict__ b_ih,
    const float* __restrict__ b_hh, const unsigned short* __restrict__ whh_bf,
    float* __restrict__ h0out) {
  __shared__ unsigned short whh_s[G3 * Hn];   // 96 KB bf16
  __shared__ unsigned short h_stage[BT * Hn]; // 16 KB bf16
  __shared__ float wih_s[G3 * 12];
  __shared__ float bih_s[G3];
  __shared__ float bhh_s[G3];
  __shared__ float xs[BT * 12];

  const int lane = threadIdx.x & 31;
  const int wave = threadIdx.x >> 5;
  const int rt = wave >> 1;       // row-tile 0..3
  const int cg = wave & 1;        // col-group 0..1
  const int hi = (lane >> 4) & 1; // +8 rows for upper half-wave
  const int ln16 = lane & 15;
  const int rowG0 = blockIdx.x * BT;

  copy_b16(whh_s, whh_bf, G3 * Hn);
  copy_f32(wih_s, W_ih, G3 * 12);
  copy_f32(bih_s, b_ih, G3);
  copy_f32(bhh_s, b_hh, G3);
  { u32x4 z = {0u,0u,0u,0u};
    u32x4* d = (u32x4*)h_stage;
    for (int i = threadIdx.x; i < (BT * Hn) / 8; i += blockDim.x) d[i] = z; }

  float y[4][8];
  #pragma unroll
  for (int j = 0; j < 4; ++j)
    #pragma unroll
    for (int v = 0; v < 8; ++v) y[j][v] = 0.f;

  for (int t = 0; t < PRE; ++t) {
    for (int q = threadIdx.x; q < BT * 12; q += blockDim.x) {
      int row = q / 12, c = q % 12;
      xs[q] = (c < KIN) ? pre_x[(t * Bn + rowG0 + row) * KIN + c]
                        : pre_y[(t * Bn + rowG0 + row) * KOUT + (c - KIN)];
    }
    __syncthreads(); // xs + prev h_stage visible

    v8f aR[4], aZ[4], aN[4];
    #pragma unroll
    for (int j = 0; j < 4; ++j) { aR[j] = vzero8(); aZ[j] = vzero8(); aN[j] = vzero8(); }

    #pragma unroll
    for (int kc = 0; kc < 4; ++kc) {
      v16bf ah = load_frag(h_stage, Hn, rt * 16, kc, lane);
      #pragma unroll
      for (int j = 0; j < 4; ++j) {
        int cR = cg * 64 + j * 16;
        aR[j] = wmma_bf16(ah, load_frag(whh_s, Hn, cR,        kc, lane), aR[j]);
        aZ[j] = wmma_bf16(ah, load_frag(whh_s, Hn, 128 + cR,  kc, lane), aZ[j]);
        aN[j] = wmma_bf16(ah, load_frag(whh_s, Hn, 256 + cR,  kc, lane), aN[j]);
      }
    }
    __syncthreads(); // GEMM reads done before h_stage rewrite

    #pragma unroll
    for (int j = 0; j < 4; ++j) {
      int col = cg * 64 + j * 16 + ln16;
      const float* wr = &wih_s[col * 12];
      const float* wz = &wih_s[(128 + col) * 12];
      const float* wn = &wih_s[(256 + col) * 12];
      #pragma unroll
      for (int v = 0; v < 8; ++v) {
        int rl = rt * 16 + hi * 8 + v;
        const float* xr = &xs[rl * 12];
        float gr = bih_s[col], gz = bih_s[128 + col], gn = bih_s[256 + col];
        #pragma unroll
        for (int c = 0; c < 12; ++c) {
          float xv = xr[c];
          gr += xv * wr[c]; gz += xv * wz[c]; gn += xv * wn[c];
        }
        float r = fast_sigmoid(gr + aR[j][v] + bhh_s[col]);
        float z = fast_sigmoid(gz + aZ[j][v] + bhh_s[128 + col]);
        float n = fast_tanh(gn + r * (aN[j][v] + bhh_s[256 + col]));
        float hv = (1.f - z) * n + z * y[j][v];
        y[j][v] = hv;
        h_stage[rl * Hn + col] = f2bf(hv);
      }
    }
    __syncthreads(); // gate-math xs reads done before next xs write
  }

  #pragma unroll
  for (int j = 0; j < 4; ++j) {
    int col = cg * 64 + j * 16 + ln16;
    #pragma unroll
    for (int v = 0; v < 8; ++v) {
      int rl = rt * 16 + hi * 8 + v;
      h0out[(rowG0 + rl) * Hn + col] = y[j][v];
    }
  }
}

// ---------------- RK4 GRU-ODE integration + output head ----------------
__global__ __launch_bounds__(256) void ode_kernel(
    const float* __restrict__ pre_x, const float* __restrict__ forward_x,
    const float* __restrict__ Mbuf, const float* __restrict__ h0in,
    const unsigned short* __restrict__ wcih_bf, const unsigned short* __restrict__ wchh_bf,
    const unsigned short* __restrict__ w1_bf,
    const float* __restrict__ W_e, const float* __restrict__ b_e,
    const float* __restrict__ bc_ih, const float* __restrict__ bc_hh,
    const float* __restrict__ W2, float* __restrict__ out) {
  __shared__ unsigned short wcih_s[G3 * Hn];      // 96 KB
  __shared__ unsigned short wchh_s[G3 * Hn];      // 96 KB
  __shared__ unsigned short w1_s[2 * Hn * Hn];    // 64 KB
  __shared__ unsigned short x_stage[BT * Hn];     // 16 KB
  __shared__ unsigned short y_stage[BT * Hn];     // 16 KB
  __shared__ float u_s[3][BT][KIN];               // u(t0), u(mid), u(t0+hs)
  __shared__ float we_s[Hn * KIN];
  __shared__ float be_s[Hn];
  __shared__ float bci_s[G3];
  __shared__ float bch_s[G3];
  __shared__ float w2_s[KOUT * 2 * Hn];
  __shared__ float out_s[BT * KOUT];

  const int lane = threadIdx.x & 31;
  const int wave = threadIdx.x >> 5;
  const int rt = wave >> 1;
  const int cg = wave & 1;
  const int hi = (lane >> 4) & 1;
  const int ln16 = lane & 15;
  const int rowG0 = blockIdx.x * BT;

  copy_b16(wcih_s, wcih_bf, G3 * Hn);
  copy_b16(wchh_s, wchh_bf, G3 * Hn);
  copy_b16(w1_s, w1_bf, 2 * Hn * Hn);
  copy_f32(we_s, W_e, Hn * KIN);
  copy_f32(be_s, b_e, Hn);
  copy_f32(bci_s, bc_ih, G3);
  copy_f32(bch_s, bc_hh, G3);
  copy_f32(w2_s, W2, KOUT * 2 * Hn);

  float ycur[4][8];
  #pragma unroll
  for (int j = 0; j < 4; ++j) {
    int col = cg * 64 + j * 16 + ln16;
    #pragma unroll
    for (int v = 0; v < 8; ++v) {
      int rl = rt * 16 + hi * 8 + v;
      ycur[j][v] = h0in[(rowG0 + rl) * Hn + col];
    }
  }
  __syncthreads();

  // One stationary-GRU f-eval: kout = (GRU(tanh(u@We^T+be), yt) - yt)/UT
  auto feval = [&](int usel, float (&yt)[4][8], float (&kout)[4][8]) {
    // stage x = tanh(u @ We^T + be) and yt as bf16
    #pragma unroll
    for (int j = 0; j < 4; ++j) {
      int col = cg * 64 + j * 16 + ln16;
      const float* w = &we_s[col * KIN];
      float bias = be_s[col];
      #pragma unroll
      for (int v = 0; v < 8; ++v) {
        int rl = rt * 16 + hi * 8 + v;
        const float* uu = &u_s[usel][rl][0];
        float a = bias;
        #pragma unroll
        for (int c = 0; c < KIN; ++c) a += uu[c] * w[c];
        x_stage[rl * Hn + col] = f2bf(fast_tanh(a));
        y_stage[rl * Hn + col] = f2bf(yt[j][v]);
      }
    }
    __syncthreads();

    v8f aR[4], aZ[4], aI[4], aH[4];
    #pragma unroll
    for (int j = 0; j < 4; ++j) { aR[j] = vzero8(); aZ[j] = vzero8(); aI[j] = vzero8(); aH[j] = vzero8(); }

    #pragma unroll
    for (int kc = 0; kc < 4; ++kc) {
      v16bf ax = load_frag(x_stage, Hn, rt * 16, kc, lane);
      v16bf ay = load_frag(y_stage, Hn, rt * 16, kc, lane);
      #pragma unroll
      for (int j = 0; j < 4; ++j) {
        int cR = cg * 64 + j * 16;
        aR[j] = wmma_bf16(ax, load_frag(wcih_s, Hn, cR,       kc, lane), aR[j]);
        aR[j] = wmma_bf16(ay, load_frag(wchh_s, Hn, cR,       kc, lane), aR[j]);
        aZ[j] = wmma_bf16(ax, load_frag(wcih_s, Hn, 128 + cR, kc, lane), aZ[j]);
        aZ[j] = wmma_bf16(ay, load_frag(wchh_s, Hn, 128 + cR, kc, lane), aZ[j]);
        aI[j] = wmma_bf16(ax, load_frag(wcih_s, Hn, 256 + cR, kc, lane), aI[j]);
        aH[j] = wmma_bf16(ay, load_frag(wchh_s, Hn, 256 + cR, kc, lane), aH[j]);
      }
    }
    __syncthreads();

    #pragma unroll
    for (int j = 0; j < 4; ++j) {
      int col = cg * 64 + j * 16 + ln16;
      #pragma unroll
      for (int v = 0; v < 8; ++v) {
        float r = fast_sigmoid(aR[j][v] + bci_s[col] + bch_s[col]);
        float z = fast_sigmoid(aZ[j][v] + bci_s[128 + col] + bch_s[128 + col]);
        float n = fast_tanh(aI[j][v] + bci_s[256 + col] + r * (aH[j][v] + bch_s[256 + col]));
        float hc = (1.f - z) * n + z * yt[j][v];
        kout[j][v] = (hc - yt[j][v]) * (1.0f / UTC);
      }
    }
  };

  for (int i = 0; i < Ln; ++i) {
    // spline values: u(t0)=ys[i], u(t0+hs)=ys[i+1],
    // u(mid) = 0.5*(ys[i]+ys[i+1]) - 0.0625*hs^2*(M[i]+M[i+1])
    for (int q = threadIdx.x; q < BT * KIN; q += blockDim.x) {
      int row = q >> 3, c = q & 7;
      int gidx = (rowG0 + row) * KIN + c;
      float y0 = (i == 0) ? pre_x[63 * Sn + gidx] : forward_x[(i - 1) * Sn + gidx];
      float y1 = forward_x[i * Sn + gidx];
      float m0 = Mbuf[i * Sn + gidx];
      float m1 = Mbuf[(i + 1) * Sn + gidx];
      u_s[0][row][c] = y0;
      u_s[2][row][c] = y1;
      u_s[1][row][c] = 0.5f * (y0 + y1) - (0.0625f * HS * HS) * (m0 + m1);
      // prefetch next step's planes into cache while this step computes
      if (i + 1 < Ln) {
        __builtin_prefetch(&forward_x[(i + 1) * Sn + gidx], 0, 1);
        __builtin_prefetch(&Mbuf[(i + 2) * Sn + gidx], 0, 1);
      }
    }
    for (int q = threadIdx.x; q < BT * KOUT; q += blockDim.x) out_s[q] = 0.f;
    __syncthreads();

    float yt[4][8], ac[4][8], kk[4][8];
    #pragma unroll
    for (int j = 0; j < 4; ++j)
      #pragma unroll
      for (int v = 0; v < 8; ++v) yt[j][v] = ycur[j][v];

    feval(0, yt, kk); // k1 at t0
    #pragma unroll
    for (int j = 0; j < 4; ++j)
      #pragma unroll
      for (int v = 0; v < 8; ++v) { ac[j][v] = kk[j][v]; yt[j][v] = ycur[j][v] + 0.5f * HS * kk[j][v]; }

    feval(1, yt, kk); // k2 at mid
    #pragma unroll
    for (int j = 0; j < 4; ++j)
      #pragma unroll
      for (int v = 0; v < 8; ++v) { ac[j][v] += 2.f * kk[j][v]; yt[j][v] = ycur[j][v] + 0.5f * HS * kk[j][v]; }

    feval(1, yt, kk); // k3 at mid
    #pragma unroll
    for (int j = 0; j < 4; ++j)
      #pragma unroll
      for (int v = 0; v < 8; ++v) { ac[j][v] += 2.f * kk[j][v]; yt[j][v] = ycur[j][v] + HS * kk[j][v]; }

    feval(2, yt, kk); // k4 at t0+hs
    #pragma unroll
    for (int j = 0; j < 4; ++j)
      #pragma unroll
      for (int v = 0; v < 8; ++v) { ac[j][v] += kk[j][v]; ycur[j][v] += (HS / 6.0f) * ac[j][v]; }

    // stage updated state, then head: out = tanh(y @ W1^T) @ W2^T
    #pragma unroll
    for (int j = 0; j < 4; ++j) {
      int col = cg * 64 + j * 16 + ln16;
      #pragma unroll
      for (int v = 0; v < 8; ++v) {
        int rl = rt * 16 + hi * 8 + v;
        y_stage[rl * Hn + col] = f2bf(ycur[j][v]);
      }
    }
    __syncthreads();

    v8f aHd[8];
    #pragma unroll
    for (int j2 = 0; j2 < 8; ++j2) aHd[j2] = vzero8();
    #pragma unroll
    for (int kc = 0; kc < 4; ++kc) {
      v16bf ay = load_frag(y_stage, Hn, rt * 16, kc, lane);
      #pragma unroll
      for (int j2 = 0; j2 < 8; ++j2)
        aHd[j2] = wmma_bf16(ay, load_frag(w1_s, Hn, cg * 128 + j2 * 16, kc, lane), aHd[j2]);
    }

    float p[8][4];
    #pragma unroll
    for (int v = 0; v < 8; ++v)
      #pragma unroll
      for (int o = 0; o < 4; ++o) p[v][o] = 0.f;
    #pragma unroll
    for (int j2 = 0; j2 < 8; ++j2) {
      int col2 = cg * 128 + j2 * 16 + ln16;
      #pragma unroll
      for (int v = 0; v < 8; ++v) {
        float th = fast_tanh(aHd[j2][v]);
        #pragma unroll
        for (int o = 0; o < 4; ++o) p[v][o] += th * w2_s[o * 256 + col2];
      }
    }
    #pragma unroll
    for (int v = 0; v < 8; ++v)
      #pragma unroll
      for (int o = 0; o < 4; ++o) {
        p[v][o] += __shfl_xor(p[v][o], 1, 32);
        p[v][o] += __shfl_xor(p[v][o], 2, 32);
        p[v][o] += __shfl_xor(p[v][o], 4, 32);
        p[v][o] += __shfl_xor(p[v][o], 8, 32);
      }
    if (ln16 == 0) {
      #pragma unroll
      for (int v = 0; v < 8; ++v) {
        int rl = rt * 16 + hi * 8 + v;
        #pragma unroll
        for (int o = 0; o < 4; ++o) atomicAdd(&out_s[rl * 4 + o], p[v][o]);
      }
    }
    __syncthreads();

    for (int q = threadIdx.x; q < BT * KOUT; q += blockDim.x)
      out[(i * Bn + rowG0 + (q >> 2)) * KOUT + (q & 3)] = out_s[q];
    __syncthreads();
  }
}

// ---------------- launch ----------------
extern "C" void kernel_launch(void* const* d_in, const int* in_sizes, int n_in,
                              void* d_out, int out_size, void* d_ws, size_t ws_size,
                              hipStream_t stream) {
  (void)in_sizes; (void)n_in; (void)out_size; (void)ws_size;
  const float* pre_x     = (const float*)d_in[0];
  const float* pre_y     = (const float*)d_in[1];
  const float* forward_x = (const float*)d_in[2];
  const float* W_ih      = (const float*)d_in[3];
  const float* W_hh      = (const float*)d_in[4];
  const float* b_ih      = (const float*)d_in[5];
  const float* b_hh      = (const float*)d_in[6];
  const float* W_e       = (const float*)d_in[7];
  const float* b_e       = (const float*)d_in[8];
  const float* Wc_ih     = (const float*)d_in[9];
  const float* Wc_hh     = (const float*)d_in[10];
  const float* bc_ih     = (const float*)d_in[11];
  const float* bc_hh     = (const float*)d_in[12];
  const float* W1        = (const float*)d_in[13];
  const float* W2        = (const float*)d_in[14];

  unsigned char* ws = (unsigned char*)d_ws;
  unsigned short* whh_bf  = (unsigned short*)(ws + 0);        // 384*128 bf16
  unsigned short* wcih_bf = (unsigned short*)(ws + 98304);
  unsigned short* wchh_bf = (unsigned short*)(ws + 196608);
  unsigned short* w1_bf   = (unsigned short*)(ws + 294912);   // 256*128 bf16
  float* h0   = (float*)(ws + 360448);                        // 4096*128 f32
  float* Mbuf = (float*)(ws + 360448 + 2097152);              // 129*32768 f32

  cvt_bf16<<<(G3 * Hn + 255) / 256, 256, 0, stream>>>(W_hh, whh_bf, G3 * Hn);
  cvt_bf16<<<(G3 * Hn + 255) / 256, 256, 0, stream>>>(Wc_ih, wcih_bf, G3 * Hn);
  cvt_bf16<<<(G3 * Hn + 255) / 256, 256, 0, stream>>>(Wc_hh, wchh_bf, G3 * Hn);
  cvt_bf16<<<(2 * Hn * Hn + 255) / 256, 256, 0, stream>>>(W1, w1_bf, 2 * Hn * Hn);

  spline_kernel<<<Sn / 256, 256, 0, stream>>>(pre_x, forward_x, Mbuf);

  encoder_kernel<<<Bn / BT, 256, 0, stream>>>(pre_x, pre_y, W_ih, b_ih, b_hh, whh_bf, h0);

  ode_kernel<<<Bn / BT, 256, 0, stream>>>(pre_x, forward_x, Mbuf, h0,
                                          wcih_bf, wchh_bf, w1_bf,
                                          W_e, b_e, bc_ih, bc_hh, W2,
                                          (float*)d_out);
}
